// OptimizedMultiHeadAttention_5257039971133
// MI455X (gfx1250) — matrix operand order repaired and checked by reading, not compiled, not run on hardware
//
#include <hip/hip_runtime.h>
#include <hip/hip_bf16.h>

// ---------------------------------------------------------------------------
// Multi-head attention (B=2, S=2048, D=1024, H=16, HD=64) for gfx1250.
// bf16 WMMA everywhere, f32 accumulate, flash-style online softmax.
// V is produced transposed [B,H,HD,S] so P*V B-fragments are contiguous.
// Attention processes 32 queries/wave to amortize K/V fragment loads.
// ---------------------------------------------------------------------------

#define BB 2
#define SS 2048
#define DD 1024
#define HH 16
#define HD 64

typedef __attribute__((ext_vector_type(16))) __bf16 v16bf;
typedef __attribute__((ext_vector_type(8)))  float  v8f;

union Frag {
    unsigned u[8];
    v16bf    v;
};

__device__ __forceinline__ unsigned short f2bf(float f) {
    unsigned u = __float_as_uint(f);
    u += 0x7FFFu + ((u >> 16) & 1u);          // round-to-nearest-even
    return (unsigned short)(u >> 16);
}
__device__ __forceinline__ float bf2f(unsigned short h) {
    return __uint_as_float(((unsigned)h) << 16);
}

__device__ __forceinline__ v8f wmma_bf16(const Frag& a, const Frag& b, v8f c) {
    return __builtin_amdgcn_wmma_f32_16x16x32_bf16(
        /*neg_a=*/false, a.v, /*neg_b=*/false, b.v,
        /*c_mod=*/(short)0, c, /*reuse_a=*/false, /*reuse_b=*/false);
}

__device__ __forceinline__ float rmax16(float x) {
#pragma unroll
    for (int m = 1; m < 16; m <<= 1) x = fmaxf(x, __shfl_xor(x, m, 32));
    return x;
}
__device__ __forceinline__ float rsum16(float x) {
#pragma unroll
    for (int m = 1; m < 16; m <<= 1) x += __shfl_xor(x, m, 32);
    return x;
}

// ---------------------------------------------------------------------------
// 1) f32 -> bf16 conversion
// ---------------------------------------------------------------------------
__global__ void mha_cvt_bf16(const float* __restrict__ in,
                             unsigned short* __restrict__ out, int n) {
    int i = blockIdx.x * 256 + threadIdx.x;
    if (i < n) out[i] = f2bf(in[i]);
}

// ---------------------------------------------------------------------------
// 2) QKV projection: out = X * W^T, bf16 in/out, f32 accumulate.
//    transposeOut==0: Out[b,h,s,d]   (Q, K — RoPE-friendly)
//    transposeOut==1: Out[b,h,d,s]   (V — contiguous P*V fragments)
//    One wave per 16x64 output tile (A fragment reused across 4 N-tiles).
// ---------------------------------------------------------------------------
__global__ void mha_gemm_qkv(const unsigned short* __restrict__ Xb,
                             const unsigned short* __restrict__ Wb,
                             unsigned short* __restrict__ Out,
                             int transposeOut) {
    const int lane = threadIdx.x;
    const int mrow = lane & 15;
    const int hh   = lane >> 4;
    const int m0   = blockIdx.x * 16;   // row tile of X (M = B*S)
    const int n0   = blockIdx.y * 64;   // one head's 64 columns

    v8f acc[4] = {};

    for (int kb = 0; kb < DD; kb += 32) {
        Frag a;
#pragma unroll
        for (int v = 0; v < 8; ++v) {
            int k0 = ((v & 4) << 2) + hh * 8 + (v & 3) * 2;
            a.u[v] = *(const unsigned*)&Xb[(size_t)(m0 + mrow) * DD + kb + k0];
        }
#pragma unroll
        for (int t = 0; t < 4; ++t) {
            Frag b;
#pragma unroll
            for (int v = 0; v < 8; ++v) {
                int kk = ((v & 4) << 2) + hh * 8 + (v & 3) * 2;
                b.u[v] = *(const unsigned*)&Wb[(size_t)(n0 + t * 16 + mrow) * DD + kb + kk];
            }
            acc[t] = wmma_bf16(a, b, acc[t]);
        }
    }

#pragma unroll
    for (int t = 0; t < 4; ++t) {
#pragma unroll
        for (int r = 0; r < 8; ++r) {
            int srow = m0 + r + 8 * hh;          // global row in [0, B*S)
            int bidx = srow >> 11;               // / S
            int s    = srow & (SS - 1);
            int n    = n0 + t * 16 + mrow;
            int h    = n >> 6;
            int d    = n & 63;
            unsigned short val = f2bf(acc[t][r]);
            if (transposeOut)
                Out[(((size_t)bidx * HH + h) * HD + d) * SS + s] = val;
            else
                Out[(((size_t)bidx * HH + h) * SS + s) * HD + d] = val;
        }
    }
}

// ---------------------------------------------------------------------------
// 3) RoPE in place on Q or K ([B,H,S,HD] bf16). One thread per (row, d<32).
// ---------------------------------------------------------------------------
__global__ void mha_rope(unsigned short* __restrict__ buf, int n) {
    int i = blockIdx.x * 256 + threadIdx.x;
    if (i >= n) return;
    int d = i & 31;
    int r = i >> 5;                 // row index in [0, B*H*S)
    int s = r & (SS - 1);           // position
    size_t base = (size_t)r * HD;
    // inv_freq = 10000^(-d/32) = exp(-d * ln(10000)/32)
    float f  = __expf(-(float)d * 0.28782313662425572f);
    float an = (float)s * f;
    float c  = __cosf(an);
    float sn = __sinf(an);
    float x1 = bf2f(buf[base + d]);
    float x2 = bf2f(buf[base + d + 32]);
    buf[base + d]      = f2bf(x1 * c - x2 * sn);
    buf[base + d + 32] = f2bf(x2 * c + x1 * sn);
}

// ---------------------------------------------------------------------------
// 4) Flash attention: one wave per (b,h, 32-query block). Key blocks of 32.
//    Two query tiles share every K-block and V-block fragment.
//    scores = Q(16x64) Kt(64x32) via 2 chained WMMAs per 16x16 tile,
//    online softmax (row stats via 16-lane shuffles), P via LDS reshape,
//    out += P(16x32) V(32x64) via 4 WMMAs per query tile.
// ---------------------------------------------------------------------------
__global__ void mha_attn(const unsigned short* __restrict__ Q,
                         const unsigned short* __restrict__ K,
                         const unsigned short* __restrict__ Vt,
                         unsigned short* __restrict__ O) {
    const int lane = threadIdx.x;
    const int mrow = lane & 15;
    const int hh   = lane >> 4;
    const int bh   = blockIdx.y;            // 0 .. B*H-1
    const int q0   = blockIdx.x * 32;

    const unsigned short* Qh  = Q  + (size_t)bh * SS * HD;
    const unsigned short* Kh  = K  + (size_t)bh * SS * HD;
    const unsigned short* Vth = Vt + (size_t)bh * HD * SS;   // [d][s]
    unsigned short*       Oh  = O  + (size_t)bh * SS * HD;

    __shared__ unsigned short plds[2][16 * 32];   // P tiles, row-major [m][t]

    // Q fragments: [query tile][d-half], resident across the key loop
    Frag aq[2][2];
#pragma unroll
    for (int qt = 0; qt < 2; ++qt) {
#pragma unroll
        for (int v = 0; v < 8; ++v) {
            int k0 = ((v & 4) << 2) + hh * 8 + (v & 3) * 2;
            size_t row = (size_t)(q0 + qt * 16 + mrow) * HD;
            aq[qt][0].u[v] = *(const unsigned*)&Qh[row + k0];
            aq[qt][1].u[v] = *(const unsigned*)&Qh[row + 32 + k0];
        }
    }

    float mrun[2][8], lrun[2][8];
    v8f acc[2][4] = {};
#pragma unroll
    for (int qt = 0; qt < 2; ++qt)
#pragma unroll
        for (int r = 0; r < 8; ++r) { mrun[qt][r] = -1e30f; lrun[qt][r] = 0.0f; }

    const float scale = 0.125f;   // 1/sqrt(64)

    for (int t0 = 0; t0 < SS; t0 += 32) {
        // ---- scores: shared K fragments across both query tiles ----
        v8f sc[2][2];
#pragma unroll
        for (int tt = 0; tt < 2; ++tt) {
            Frag bk0, bk1;
            int t = t0 + tt * 16 + mrow;
#pragma unroll
            for (int v = 0; v < 8; ++v) {
                int kk = ((v & 4) << 2) + hh * 8 + (v & 3) * 2;
                bk0.u[v] = *(const unsigned*)&Kh[(size_t)t * HD + kk];
                bk1.u[v] = *(const unsigned*)&Kh[(size_t)t * HD + 32 + kk];
            }
#pragma unroll
            for (int qt = 0; qt < 2; ++qt) {
                v8f s = {};
                s = wmma_bf16(aq[qt][0], bk0, s);
                s = wmma_bf16(aq[qt][1], bk1, s);
                sc[qt][tt] = s;
            }
        }

        // ---- online softmax update + stage P into LDS ----
#pragma unroll
        for (int qt = 0; qt < 2; ++qt) {
#pragma unroll
            for (int r = 0; r < 8; ++r) {
                float slo = sc[qt][0][r] * scale;
                float shi = sc[qt][1][r] * scale;
                float rx  = rmax16(fmaxf(slo, shi));
                float mn  = fmaxf(mrun[qt][r], rx);
                float cor = __expf(mrun[qt][r] - mn);
                float pl  = __expf(slo - mn);
                float ph  = __expf(shi - mn);
                lrun[qt][r] = lrun[qt][r] * cor + rsum16(pl + ph);
                mrun[qt][r] = mn;
#pragma unroll
                for (int dt = 0; dt < 4; ++dt) acc[qt][dt][r] *= cor;
                plds[qt][(r + 8 * hh) * 32 + mrow]      = f2bf(pl);
                plds[qt][(r + 8 * hh) * 32 + 16 + mrow] = f2bf(ph);
            }
        }
        __syncthreads();

        // ---- reload P in A-layout ----
        Frag ap[2];
#pragma unroll
        for (int qt = 0; qt < 2; ++qt)
#pragma unroll
            for (int v = 0; v < 8; ++v) {
                int k0 = ((v & 4) << 2) + hh * 8 + (v & 3) * 2;
                ap[qt].u[v] = *(const unsigned*)&plds[qt][mrow * 32 + k0];
            }

        // ---- out += P * V : shared V fragments across both query tiles ----
#pragma unroll
        for (int dt = 0; dt < 4; ++dt) {
            Frag bv;
            int d = dt * 16 + mrow;
#pragma unroll
            for (int v = 0; v < 8; ++v) {
                // Vt[d][t], pairs along t are contiguous
                bv.u[v] = *(const unsigned*)&Vth[(size_t)d * SS + t0 + ((v & 4) << 2) + hh * 8 + (v & 3) * 2];
            }
            acc[0][dt] = wmma_bf16(ap[0], bv, acc[0][dt]);
            acc[1][dt] = wmma_bf16(ap[1], bv, acc[1][dt]);
        }
        __syncthreads();   // protect plds before next iteration's writes
    }

    // ---- normalize and store bf16 ----
#pragma unroll
    for (int qt = 0; qt < 2; ++qt)
#pragma unroll
        for (int dt = 0; dt < 4; ++dt)
#pragma unroll
            for (int r = 0; r < 8; ++r) {
                int srow = q0 + qt * 16 + r + 8 * hh;
                float o  = acc[qt][dt][r] / lrun[qt][r];
                Oh[(size_t)srow * HD + dt * 16 + mrow] = f2bf(o);
            }
}

// ---------------------------------------------------------------------------
// 5) Output projection: d_out[m, n] = sum_k Attn[m, k] * Wo[n, k]  (f32 out)
//    Attn stored head-major [B,H,S,HD]; logical k = h*64 + d.
// ---------------------------------------------------------------------------
__global__ void mha_gemm_out(const unsigned short* __restrict__ Attn,
                             const unsigned short* __restrict__ Wob,
                             float* __restrict__ Out) {
    const int lane = threadIdx.x;
    const int mrow = lane & 15;
    const int hh   = lane >> 4;
    const int m0   = blockIdx.x * 16;
    const int n0   = blockIdx.y * 64;

    const int srow = m0 + mrow;
    const int bidx = srow >> 11;
    const int s    = srow & (SS - 1);

    v8f acc[4] = {};

    for (int kb = 0; kb < DD; kb += 32) {
        const int h    = kb >> 6;          // head for this k-block
        const int dofs = kb & 63;          // 0 or 32
        Frag a;
#pragma unroll
        for (int v = 0; v < 8; ++v) {
            int k0 = ((v & 4) << 2) + hh * 8 + (v & 3) * 2;
            a.u[v] = *(const unsigned*)
                &Attn[(((size_t)bidx * HH + h) * SS + s) * HD + dofs + k0];
        }
#pragma unroll
        for (int t = 0; t < 4; ++t) {
            Frag b;
#pragma unroll
            for (int v = 0; v < 8; ++v) {
                int kk = ((v & 4) << 2) + hh * 8 + (v & 3) * 2;
                b.u[v] = *(const unsigned*)&Wob[(size_t)(n0 + t * 16 + mrow) * DD + kb + kk];
            }
            acc[t] = wmma_bf16(a, b, acc[t]);
        }
    }

#pragma unroll
    for (int t = 0; t < 4; ++t) {
#pragma unroll
        for (int r = 0; r < 8; ++r) {
            int row = m0 + r + 8 * hh;
            int n   = n0 + t * 16 + mrow;
            Out[(size_t)row * DD + n] = acc[t][r];
        }
    }
}

// ---------------------------------------------------------------------------
// Launch
// ---------------------------------------------------------------------------
extern "C" void kernel_launch(void* const* d_in, const int* in_sizes, int n_in,
                              void* d_out, int out_size, void* d_ws, size_t ws_size,
                              hipStream_t stream) {
    const float* x  = (const float*)d_in[0];
    const float* Wq = (const float*)d_in[1];
    const float* Wk = (const float*)d_in[2];
    const float* Wv = (const float*)d_in[3];
    const float* Wo = (const float*)d_in[4];
    float* out = (float*)d_out;

    const size_t nX = (size_t)BB * SS * DD;         // 4,194,304
    const size_t nW = (size_t)DD * DD;              // 1,048,576
    const size_t nQ = (size_t)BB * HH * SS * HD;    // 4,194,304

    unsigned char* ws = (unsigned char*)d_ws;
    unsigned short* xb  = (unsigned short*)(ws);                       size_t off = nX * 2;
    unsigned short* wqb = (unsigned short*)(ws + off); off += nW * 2;
    unsigned short* wkb = (unsigned short*)(ws + off); off += nW * 2;
    unsigned short* wvb = (unsigned short*)(ws + off); off += nW * 2;
    unsigned short* wob = (unsigned short*)(ws + off); off += nW * 2;
    unsigned short* Qb  = (unsigned short*)(ws + off); off += nQ * 2;
    unsigned short* Kb  = (unsigned short*)(ws + off); off += nQ * 2;
    unsigned short* Vtb = (unsigned short*)(ws + off); off += nQ * 2;  // [B,H,HD,S]
    unsigned short* Ab  = (unsigned short*)(ws + off); off += nQ * 2;
    (void)ws_size; (void)in_sizes; (void)n_in; (void)out_size;

    // 1) convert inputs to bf16
    mha_cvt_bf16<<<dim3((unsigned)((nX + 255) / 256)), 256, 0, stream>>>(x,  xb,  (int)nX);
    mha_cvt_bf16<<<dim3((unsigned)((nW + 255) / 256)), 256, 0, stream>>>(Wq, wqb, (int)nW);
    mha_cvt_bf16<<<dim3((unsigned)((nW + 255) / 256)), 256, 0, stream>>>(Wk, wkb, (int)nW);
    mha_cvt_bf16<<<dim3((unsigned)((nW + 255) / 256)), 256, 0, stream>>>(Wv, wvb, (int)nW);
    mha_cvt_bf16<<<dim3((unsigned)((nW + 255) / 256)), 256, 0, stream>>>(Wo, wob, (int)nW);

    // 2) Q/K/V projections (one wave per 16x64 tile); V stored transposed
    dim3 ggemm(BB * SS / 16, DD / 64);   // (256, 16)
    mha_gemm_qkv<<<ggemm, 32, 0, stream>>>(xb, wqb, Qb, 0);
    mha_gemm_qkv<<<ggemm, 32, 0, stream>>>(xb, wkb, Kb, 0);
    mha_gemm_qkv<<<ggemm, 32, 0, stream>>>(xb, wvb, Vtb, 1);

    // 3) RoPE on Q and K
    const int nRope = BB * HH * SS * (HD / 2);       // 2,097,152
    mha_rope<<<dim3(nRope / 256), 256, 0, stream>>>(Qb, nRope);
    mha_rope<<<dim3(nRope / 256), 256, 0, stream>>>(Kb, nRope);

    // 4) flash attention: one wave per (b,h, 32-query block)
    dim3 gattn(SS / 32, BB * HH);        // (64, 32)
    mha_attn<<<gattn, 32, 0, stream>>>(Qb, Kb, Vtb, Ab);

    // 5) output projection -> f32
    mha_gemm_out<<<ggemm, 32, 0, stream>>>(Ab, wob, out);
}
